// SE3TransformerIt_43516608643527
// MI455X (gfx1250) — compile-verified, matrix operand order checked by reading, not run on hardware
//
#include <hip/hip_runtime.h>
#include <hip/hip_bf16.h>
#include <math.h>

// =====================================================================
// SE(3)-Transformer forward for MI455X (gfx1250, wave32).
//
//  * Hot FLOPs = edge-batched small GEMMs (radial MLP E x 65 x 32 x R, and
//    h0-gather x concatenated path weights) -> V_WMMA_F32_16X16X32_F16,
//    16-edge tiles per wave32, f16 operands, f32 accumulate.
//  * Weight B-fragments are pre-packed once per call into d_ws in the exact
//    per-lane WMMA layout, so the edge kernel loads each fragment as one
//    32B vector load (L2/WGP$-resident) instead of 16 scattered b32 + cvts.
//  * Degree-1/2 feature gathers (raw f32, no conversion) are staged with
//    GLOBAL_LOAD_ASYNC_TO_LDS_B32 (ASYNCcnt / s_wait_asynccnt 0), per
//    CDNA5 ISA. EXEC is all-ones in those loops.
//  * Memory floor ~70-90us @ 23.3 TB/s (w: 67MB, per-edge msgs: 105MB per
//    attention layer, 1 write + 1 read each). Softmax/scatter use native
//    global f32 atomics over 8192x{4,100} cells.
//  * Param flattening assumed = JAX tree-leaf order (sorted dict keys).
//  * Scratch ~197 MB of d_ws; everything read is initialized each call.
// =====================================================================

#define DEV __device__ __forceinline__

typedef __attribute__((ext_vector_type(16))) _Float16 v16h;
typedef __attribute__((ext_vector_type(8)))  float    v8f;

constexpr int NN   = 8192;       // nodes
constexpr int NE   = 262144;     // edges
constexpr int NGR  = 4;
constexpr int NPGc = 2048;
constexpr float EPSV = 1e-6f;

// h storage strides (fixed, covering max dims across layers)
constexpr int H0S = 64;          // lo=0: up to 64 ch x 1 comp
constexpr int H1S = 48;          // lo=1: up to 16 ch x 3 comp
constexpr int H2S = 20;          // lo=2: up to  4 ch x 5 comp

constexpr int RPADMAX  = 208;    // R max 196 -> 13 tiles of 16
constexpr int SCPADMAX = 80;     // SC max 66 -> 5 tiles of 16
constexpr int MSGW     = 100;    // attn msg floats/edge: 32 + 16*3 + 4*5

// ---------------- WMMA helpers ----------------
DEV v8f wmma16(v16h a, v16h b, v8f c) {
  return __builtin_amdgcn_wmma_f32_16x16x32_f16(false, a, false, b, (short)0, c,
                                                false, false);
}
DEV int kmap(int j, int h) { return ((j >> 2) << 4) + (h << 3) + ((j & 3) << 1); }

// A tile (16 rows x 32 K) from LDS f16, row-major, row stride ld
DEV v16h load_a_lds(const _Float16* base, int ld, int koff, int lane) {
  int r = lane & 15, h = lane >> 4;
  v16h a;
#pragma unroll
  for (int j = 0; j < 8; ++j) {
    int k0 = kmap(j, h) + koff;
    a[2 * j]     = base[r * ld + k0];
    a[2 * j + 1] = base[r * ld + k0 + 1];
  }
  return a;
}
// B tile (32 K x 16 cols) from global f32 (kmax x nmax), zero-padded
DEV v16h load_b_glb(const float* B, int ld, int kmax, int nmax, int koff,
                    int noff, int lane) {
  int r = lane & 15, h = lane >> 4;
  int n = noff + r;
  v16h b;
#pragma unroll
  for (int j = 0; j < 8; ++j) {
    int k0 = kmap(j, h) + koff;
    float v0 = (k0 < kmax && n < nmax) ? B[k0 * ld + n] : 0.f;
    float v1 = (k0 + 1 < kmax && n < nmax) ? B[(k0 + 1) * ld + n] : 0.f;
    b[2 * j]     = (_Float16)v0;
    b[2 * j + 1] = (_Float16)v1;
  }
  return b;
}
// B tile for concatenated transposed path weights [W00^T | W01^T | W02^T]
DEV float wcat_elem(const float* W00, const float* W01, const float* W02,
                    int fin0, int mo0, int mo1, int mo2, int k, int c) {
  if (k >= fin0) return 0.f;
  if (c < mo0) return W00[c * fin0 + k];
  c -= mo0;
  if (c < mo1) return W01[c * fin0 + k];
  c -= mo1;
  if (c < mo2 && W02) return W02[c * fin0 + k];
  return 0.f;
}
DEV v16h load_b_wcat(const float* W00, const float* W01, const float* W02,
                     int fin0, int mo0, int mo1, int mo2, int koff, int noff,
                     int lane) {
  int r = lane & 15, h = lane >> 4;
  int n = noff + r;
  v16h b;
#pragma unroll
  for (int j = 0; j < 8; ++j) {
    int k0 = kmap(j, h) + koff;
    b[2 * j]     = (_Float16)wcat_elem(W00, W01, W02, fin0, mo0, mo1, mo2, k0, n);
    b[2 * j + 1] = (_Float16)wcat_elem(W00, W01, W02, fin0, mo0, mo1, mo2, k0 + 1, n);
  }
  return b;
}
// pre-packed fragment fetch: one aligned 32B vector load per lane
DEV v16h bfrag(const _Float16* pk, int tile, int lane) {
  return ((const v16h*)pk)[tile * 32 + lane];
}

// ---- CDNA5 async global->LDS copy (ASYNCcnt) ----
// Generic pointers to LDS carry the LDS byte offset in their low 32 bits
// (ISA 10.2 aperture rule).
DEV void async_ld_b32(void* ldsp, const void* g) {
  unsigned loff = (unsigned)(unsigned long long)ldsp;
  unsigned long long ga = (unsigned long long)g;
  asm volatile("global_load_async_to_lds_b32 %0, %1, off"
               :: "v"(loff), "v"(ga) : "memory");
}
DEV void async_wait0() {
  asm volatile("s_wait_asynccnt 0x0" ::: "memory");
}

DEV void atomicMaxF(float* addr, float val) {
  unsigned int* ua = (unsigned int*)addr;
  unsigned int old = *ua;
  while (__uint_as_float(old) < val) {
    unsigned int assumed = old;
    old = atomicCAS(ua, assumed, __float_as_uint(val));
    if (old == assumed) break;
  }
}

// ---------------- weight fragment pre-pack ----------------
struct PackCfg {
  const float* B;                       // mode 0
  const float *W00, *W01, *W02;         // mode 1
  int ld, kmax, nmax;                   // mode 0
  int fin0, mo0, mo1, mo2;              // mode 1
  int ntiles, kchunks;
  _Float16* out;                        // [ntiles*kchunks][32 lanes][16 f16]
  int mode;
};
__global__ void pack_kernel(PackCfg p) {
  int t = blockIdx.x * blockDim.x + threadIdx.x;
  int total = p.ntiles * p.kchunks * 32;
  if (t >= total) return;
  int lane = t & 31;
  int tile = t >> 5;
  int kc = tile % p.kchunks, nt = tile / p.kchunks;
  v16h b = (p.mode == 0)
               ? load_b_glb(p.B, p.ld, p.kmax, p.nmax, kc * 32, nt * 16, lane)
               : load_b_wcat(p.W00, p.W01, p.W02, p.fin0, p.mo0, p.mo1, p.mo2,
                             kc * 32, nt * 16, lane);
  ((v16h*)p.out)[t] = b;
}

// ---------------- utility kernels ----------------
__global__ void fill_kernel(float* p, int n, float v) {
  int t = blockIdx.x * blockDim.x + threadIdx.x;
  if (t < n) p[t] = v;
}

__global__ void init_kernel(const int* s, const float* x, const float* emb,
                            float* h0, float* h1, float* h2, float* xbuf) {
  int n = blockIdx.x * blockDim.x + threadIdx.x;
  if (n >= NN) return;
  int si = s[n];
  for (int i = 0; i < 64; ++i) h0[n * H0S + i] = emb[si * 64 + i];
  for (int i = 0; i < H1S; ++i) h1[n * H1S + i] = 0.f;
  for (int i = 0; i < H2S; ++i) h2[n * H2S + i] = 0.f;
  for (int m = 0; m < 3; ++m) xbuf[n * 3 + m] = x[n * 3 + m];
}

__global__ void w_kernel(const int* src, const int* dst, const float* f, float* w) {
  int t = blockIdx.x * blockDim.x + threadIdx.x;
  if (t >= NE * 64) return;
  int e = t >> 6, i = t & 63;
  w[t] = f[src[e] * H0S + i] * f[dst[e] * H0S + i];
}

__global__ void geom_kernel(const int* src, const int* dst, const float* xbuf,
                            float* rlen, float* Y1, float* Y2) {
  int e = blockIdx.x * blockDim.x + threadIdx.x;
  if (e >= NE) return;
  int sn = src[e], dn = dst[e];
  float dx = xbuf[dn * 3 + 0] - xbuf[sn * 3 + 0];
  float dy = xbuf[dn * 3 + 1] - xbuf[sn * 3 + 1];
  float dz = xbuf[dn * 3 + 2] - xbuf[sn * 3 + 2];
  float r = sqrtf(dx * dx + dy * dy + dz * dz + EPSV);
  rlen[e] = r;
  float ux = dx / r, uy = dy / r, uz = dz / r;
  const float c1 = 0.4886025119f;
  Y1[e * 3 + 0] = c1 * uy;
  Y1[e * 3 + 1] = c1 * uz;
  Y1[e * 3 + 2] = c1 * ux;
  const float c2 = 1.0925484306f, c20 = 0.3153915653f, c22 = 0.5462742153f;
  Y2[e * 5 + 0] = c2 * ux * uy;
  Y2[e * 5 + 1] = c2 * uy * uz;
  Y2[e * 5 + 2] = c20 * (3.f * uz * uz - 1.f);
  Y2[e * 5 + 3] = c2 * ux * uz;
  Y2[e * 5 + 4] = c22 * (ux * ux - uy * uy);
}

__global__ void q_kernel(const float* h0, const float* Wq, int fin0, float* qnode) {
  int t = blockIdx.x * blockDim.x + threadIdx.x;
  if (t >= NN * 32) return;
  int n = t >> 5, o = t & 31;
  float acc = 0.f;
  for (int i = 0; i < fin0; ++i) acc += Wq[o * fin0 + i] * h0[n * H0S + i];
  qnode[t] = acc;
}

// ---------------- edge mega-kernel ----------------
struct EdgeCfg {
  const int *src, *dst;
  const float *w, *rlen, *Y1, *Y2;
  const float *h0, *h1, *h2;
  const float *qnode;
  const float *rb1, *rb2;
  const _Float16 *rw1p, *rw2p, *scp;   // pre-packed B fragments
  const float *W10, *W11, *W20, *W22;  // small VALU path weights
  float *msg, *logits, *nodeMax;       // attention path
  float *agg0, *agg1;                  // gconv direct-scatter path
  int fin0, C1, C2;                    // input rep dims
  int mo0, mo1, mo2;                   // output rep dims
  int R, SC;
  int off00, off10, off20, off01, off11, off02, off22;
  int attn;
};

__global__ __launch_bounds__(64) void edge_kernel(EdgeCfg c) {
  __shared__ _Float16 s_feat[2][16 * 96];
  __shared__ _Float16 s_h0[2][16 * 64];
  __shared__ _Float16 s_hid[2][16 * 32];
  __shared__ float    s_h1[2][16 * H1S];
  __shared__ float    s_h2[2][16 * H2S];
  __shared__ float    s_rout[2][16 * RPADMAX];
  __shared__ float    s_sc[2][16 * SCPADMAX];

  int tid = threadIdx.x;
  int wv = tid >> 5, lane = tid & 31;       // wave32
  int ebase = blockIdx.x * 32 + wv * 16;

  // ---- stage per-tile data into LDS ----
  // raw f32 gathers: CDNA5 async global->LDS engine (trip counts are
  // multiples of 32 -> EXEC all-ones on every issue)
  for (int idx = lane; idx < 16 * c.C1 * 3; idx += 32) {
    int e = idx / (c.C1 * 3), j = idx % (c.C1 * 3);
    int sn = c.src[ebase + e];
    async_ld_b32(&s_h1[wv][e * H1S + j], &c.h1[sn * H1S + j]);
  }
  if (c.C2 > 0) {
    for (int idx = lane; idx < 16 * c.C2 * 5; idx += 32) {
      int e = idx / (c.C2 * 5), j = idx % (c.C2 * 5);
      int sn = c.src[ebase + e];
      async_ld_b32(&s_h2[wv][e * H2S + j], &c.h2[sn * H2S + j]);
    }
  }
  // f32 -> f16 staging (needs conversion, regular path)
  for (int idx = lane; idx < 16 * 96; idx += 32) {
    int e = idx / 96, k = idx % 96;
    int edge = ebase + e;
    float v = 0.f;
    if (k == 0) v = c.rlen[edge];
    else if (k < 65) v = c.w[edge * 64 + (k - 1)];
    s_feat[wv][idx] = (_Float16)v;
  }
  for (int idx = lane; idx < 16 * c.fin0; idx += 32) {
    int e = idx / c.fin0, i = idx % c.fin0;
    int sn = c.src[ebase + e];
    s_h0[wv][e * 64 + i] = (_Float16)c.h0[sn * H0S + i];
  }
  async_wait0();
  __syncthreads();

  // ---- radial MLP stage 1: hid = relu(feat @ rw1 + rb1) (16 x 32) ----
  {
    int r = lane & 15, h = lane >> 4;
#pragma unroll
    for (int nt = 0; nt < 2; ++nt) {
      v8f acc = {};
#pragma unroll
      for (int kc = 0; kc < 3; ++kc) {
        v16h a = load_a_lds(s_feat[wv], 96, kc * 32, lane);
        v16h b = bfrag(c.rw1p, nt * 3 + kc, lane);
        acc = wmma16(a, b, acc);
      }
      int col = nt * 16 + r;
      float bias = c.rb1[col];
#pragma unroll
      for (int i = 0; i < 8; ++i) {
        int m = i + h * 8;
        s_hid[wv][m * 32 + col] = (_Float16)fmaxf(acc[i] + bias, 0.f);
      }
    }
  }
  __syncthreads();

  // ---- stage 2: rout = hid @ rw2 + rb2 ;  sc = h0s @ [W00^T|W01^T|W02^T] ----
  {
    int r = lane & 15, h = lane >> 4;
    int Rtiles = (c.R + 15) >> 4;
    for (int nt = 0; nt < Rtiles; ++nt) {
      v8f acc = {};
      v16h a = load_a_lds(s_hid[wv], 32, 0, lane);
      v16h b = bfrag(c.rw2p, nt, lane);
      acc = wmma16(a, b, acc);
      int col = nt * 16 + r;
      float bias = (col < c.R) ? c.rb2[col] : 0.f;
#pragma unroll
      for (int i = 0; i < 8; ++i)
        s_rout[wv][(i + h * 8) * RPADMAX + col] = acc[i] + bias;
    }
    int SCtiles = (c.SC + 15) >> 4;
    int kch = c.fin0 >> 5;
    for (int nt = 0; nt < SCtiles; ++nt) {
      v8f acc = {};
      for (int kc = 0; kc < kch; ++kc) {
        v16h a = load_a_lds(s_h0[wv], 64, kc * 32, lane);
        v16h b = bfrag(c.scp, nt * kch + kc, lane);
        acc = wmma16(a, b, acc);
      }
      int col = nt * 16 + r;
#pragma unroll
      for (int i = 0; i < 8; ++i)
        s_sc[wv][(i + h * 8) * SCPADMAX + col] = acc[i];
    }
  }
  __syncthreads();

  // ---- VALU: assemble equivariant messages; logits / direct scatter ----
  {
    int e = lane & 15, hh = lane >> 4;   // 2 lanes share one edge
    int edge = ebase + e;
    int dn = c.dst[edge];
    const float* rt = &s_rout[wv][e * RPADMAX];
    const float* scv = &s_sc[wv][e * SCPADMAX];
    float Y1e[3];
#pragma unroll
    for (int m = 0; m < 3; ++m) Y1e[m] = c.Y1[edge * 3 + m];
    float Y2e[5] = {0, 0, 0, 0, 0};
    if (c.C2 > 0 || c.mo2 > 0) {
#pragma unroll
      for (int m = 0; m < 5; ++m) Y2e[m] = c.Y2[edge * 5 + m];
    }
    float inv1[16];
    for (int i = 0; i < c.C1; ++i) {
      float t = 0.f;
      for (int m = 0; m < 3; ++m) t += s_h1[wv][e * H1S + i * 3 + m] * Y1e[m];
      inv1[i] = t;
    }
    float inv2[4];
    if (c.C2 > 0)
      for (int i = 0; i < c.C2; ++i) {
        float t = 0.f;
        for (int m = 0; m < 5; ++m) t += s_h2[wv][e * H2S + i * 5 + m] * Y2e[m];
        inv2[i] = t;
      }

    // lo = 0  (also attention keys)
    float lg[2] = {0.f, 0.f};
    int half0 = c.mo0 >> 1;
    for (int o = hh * half0; o < hh * half0 + half0; ++o) {
      float acc = rt[c.off00 + o] * scv[o];
      float t = 0.f;
      for (int i = 0; i < c.C1; ++i) t += c.W10[o * c.C1 + i] * inv1[i];
      acc += rt[c.off10 + o] * t;
      if (c.C2 > 0) {
        t = 0.f;
        for (int i = 0; i < c.C2; ++i) t += c.W20[o * c.C2 + i] * inv2[i];
        acc += rt[c.off20 + o] * t;
      }
      if (c.attn) {
        c.msg[edge * MSGW + o] = acc;
        lg[(o >> 3) & 1] += c.qnode[dn * 32 + o] * acc;
      } else {
        atomicAdd(&c.agg0[dn * H0S + o], acc);
      }
    }
    if (c.attn) {
#pragma unroll
      for (int t2 = 0; t2 < 2; ++t2) {
        int head = hh * 2 + t2;
        float lv = lg[t2] * 0.35355339059327373f;   // 1/sqrt(DK=8)
        c.logits[edge * 4 + head] = lv;
        atomicMaxF(&c.nodeMax[dn * 4 + head], lv);
      }
    }

    // lo = 1
    int half1 = c.mo1 >> 1;
    for (int o = hh * half1; o < hh * half1 + half1; ++o) {
      float b01 = rt[c.off01 + o] * scv[c.mo0 + o];
      for (int mm = 0; mm < 3; ++mm) {
        float t = 0.f;
        for (int i = 0; i < c.C1; ++i)
          t += c.W11[o * c.C1 + i] * s_h1[wv][e * H1S + i * 3 + mm];
        float v = b01 * Y1e[mm] + rt[c.off11 + o] * t;
        if (c.attn) c.msg[edge * MSGW + 32 + o * 3 + mm] = v;
        else atomicAdd(&c.agg1[dn * H1S + o * 3 + mm], v);
      }
    }

    // lo = 2 (attention layers only)
    if (c.mo2 > 0) {
      int half2 = c.mo2 >> 1;
      for (int o = hh * half2; o < hh * half2 + half2; ++o) {
        float b02 = rt[c.off02 + o] * scv[c.mo0 + c.mo1 + o];
        for (int mm = 0; mm < 5; ++mm) {
          float v = b02 * Y2e[mm];
          if (c.off22 >= 0) {
            float t = 0.f;
            for (int i = 0; i < c.C2; ++i)
              t += c.W22[o * c.C2 + i] * s_h2[wv][e * H2S + i * 5 + mm];
            v += rt[c.off22 + o] * t;
          }
          c.msg[edge * MSGW + 80 + o * 5 + mm] = v;
        }
      }
    }
  }
}

// ---------------- attention softmax + scatter ----------------
__global__ void den_kernel(const int* dst, const float* logits,
                           const float* nodeMax, float* den) {
  int t = blockIdx.x * blockDim.x + threadIdx.x;
  if (t >= NE * 4) return;
  int e = t >> 2, h = t & 3;
  int dn = dst[e];
  atomicAdd(&den[dn * 4 + h], expf(logits[t] - nodeMax[dn * 4 + h]));
}

__global__ void attn_scatter_kernel(const int* dst, const float* logits,
                                    const float* nodeMax, const float* den,
                                    const float* msg, float* agg0, float* agg1,
                                    float* agg2) {
  int t = blockIdx.x * blockDim.x + threadIdx.x;
  if (t >= NE * 4) return;
  int e = t >> 2, h = t & 3;
  int dn = dst[e];
  float alpha = expf(logits[t] - nodeMax[dn * 4 + h]) / (den[dn * 4 + h] + EPSV);
  const float* m = msg + e * MSGW;
  for (int j = 0; j < 8; ++j) {               // lo=0: 8 ch per head
    int o = h * 8 + j;
    atomicAdd(&agg0[dn * H0S + o], alpha * m[o]);
  }
  for (int j = 0; j < 4; ++j) {               // lo=1: 4 ch per head x 3 comps
    int o = h * 4 + j;
    for (int mm = 0; mm < 3; ++mm)
      atomicAdd(&agg1[dn * H1S + o * 3 + mm], alpha * m[32 + o * 3 + mm]);
  }
  {                                           // lo=2: 1 ch per head x 5 comps
    int o = h;
    for (int mm = 0; mm < 5; ++mm)
      atomicAdd(&agg2[dn * H2S + o * 5 + mm], alpha * m[80 + o * 5 + mm]);
  }
}

// ---------------- node kernels ----------------
struct NodeCfg {
  const float *agg0, *agg1, *agg2;
  const float *h0, *h1, *h2;
  const float *S0, *S1, *S2;
  const float *a0, *b0, *a1, *b1, *a2, *b2;
  float *o0, *o1, *o2;
  int fin0, C1, C2;
};

__global__ void node_attn_kernel(NodeCfg c) {
  int n = blockIdx.x * blockDim.x + threadIdx.x;
  if (n >= NN) return;
  for (int o = 0; o < 32; ++o) {
    float v = c.agg0[n * H0S + o];
    for (int i = 0; i < c.fin0; ++i) v += c.S0[o * c.fin0 + i] * c.h0[n * H0S + i];
    float nr = sqrtf(v * v + EPSV);
    float phi = fmaxf(c.a0[o] * nr + c.b0[o], 0.f);
    c.o0[n * H0S + o] = phi * v / nr;
  }
  for (int o = 0; o < 16; ++o) {
    float vv[3];
    for (int m = 0; m < 3; ++m) {
      float v = c.agg1[n * H1S + o * 3 + m];
      for (int i = 0; i < c.C1; ++i)
        v += c.S1[o * c.C1 + i] * c.h1[n * H1S + i * 3 + m];
      vv[m] = v;
    }
    float nr = sqrtf(vv[0] * vv[0] + vv[1] * vv[1] + vv[2] * vv[2] + EPSV);
    float phi = fmaxf(c.a1[o] * nr + c.b1[o], 0.f);
    for (int m = 0; m < 3; ++m) c.o1[n * H1S + o * 3 + m] = phi * vv[m] / nr;
  }
  for (int o = 0; o < 4; ++o) {
    float vv[5], ss = 0.f;
    for (int m = 0; m < 5; ++m) {
      float v = c.agg2[n * H2S + o * 5 + m];
      if (c.S2)
        for (int i = 0; i < c.C2; ++i)
          v += c.S2[o * c.C2 + i] * c.h2[n * H2S + i * 5 + m];
      vv[m] = v;
      ss += v * v;
    }
    float nr = sqrtf(ss + EPSV);
    float phi = fmaxf(c.a2[o] * nr + c.b2[o], 0.f);
    for (int m = 0; m < 5; ++m) c.o2[n * H2S + o * 5 + m] = phi * vv[m] / nr;
  }
}

__global__ void node_final_kernel(const float* agg0, const float* agg1,
                                  const float* h0, const float* h1,
                                  const float* Ws0, const float* Ws1, float* o0,
                                  float* o1, float* xbuf) {
  int n = blockIdx.x * blockDim.x + threadIdx.x;
  if (n >= NN) return;
  for (int o = 0; o < 64; ++o) {
    float v = agg0[n * H0S + o];
    for (int i = 0; i < 32; ++i) v += Ws0[o * 32 + i] * h0[n * H0S + i];
    o0[n * H0S + o] = v;
  }
  float v1[2][3];
  for (int o = 0; o < 2; ++o)
    for (int m = 0; m < 3; ++m) {
      float v = agg1[n * H1S + o * 3 + m];
      for (int i = 0; i < 16; ++i) v += Ws1[o * 16 + i] * h1[n * H1S + i * 3 + m];
      v1[o][m] = v;
    }
  for (int o = 0; o < 2; ++o)
    for (int m = 0; m < 3; ++m) o1[n * H1S + o * 3 + m] = v1[o][m];
  for (int m = 0; m < 3; ++m) xbuf[n * 3 + m] += v1[0][m];
}

// ---------------- outputs ----------------
__global__ void out_kernel(const float* h1, const float* target, float* out,
                           float* lossAcc) {
  int n = blockIdx.x * blockDim.x + threadIdx.x;
  if (n >= NN) return;
  int g = n >> 11, p = n & 2047;
  float v0[3], v1[3];
  for (int m = 0; m < 3; ++m) {
    v0[m] = h1[n * H1S + 0 + m];
    v1[m] = h1[n * H1S + 3 + m];
  }
  float sq = 0.f;
  int base = g * (NPGc * 9) + p * 9;
  for (int j = 0; j < 9; ++j) {
    float val = (j < 3) ? v0[j] : (j < 6) ? v1[j - 3] : v0[j - 6] + v1[j - 6];
    out[base + j] = val;
    float df = val - target[base + j];
    sq += df * df;
  }
  atomicAdd(&lossAcc[g], sq);
}

__global__ void loss_kernel(const float* lossAcc, float* out) {
  int g = threadIdx.x;
  if (g < NGR) out[NN * 9 + g] = sqrtf(lossAcc[g] / (3.f * NPGc));
}

// ---------------- parameter leaf indices (JAX tree order) ----------------
enum {
  P_F_W00 = 6, P_F_W01, P_F_W10, P_F_W11, P_F_W20, P_F_Ws0, P_F_Ws1,
  P_F_rb1, P_F_rb2, P_F_rw1, P_F_rw2,                              // 16
  P_L0_S0 = 17, P_L0_S1, P_L0_W00, P_L0_W01, P_L0_W02, P_L0_W10, P_L0_W11,
  P_L0_Wq, P_L0_a0, P_L0_a1, P_L0_a2, P_L0_b0, P_L0_b1, P_L0_b2,
  P_L0_rb1, P_L0_rb2, P_L0_rw1, P_L0_rw2,                          // 34
  P_L1_S0 = 35, P_L1_S1, P_L1_S2, P_L1_W00, P_L1_W01, P_L1_W02, P_L1_W10,
  P_L1_W11, P_L1_W20, P_L1_W22, P_L1_Wq, P_L1_a0, P_L1_a1, P_L1_a2,
  P_L1_b0, P_L1_b1, P_L1_b2, P_L1_rb1, P_L1_rb2, P_L1_rw1, P_L1_rw2  // 55
};

extern "C" void kernel_launch(void* const* d_in, const int* in_sizes, int n_in,
                              void* d_out, int out_size, void* d_ws,
                              size_t ws_size, hipStream_t stream) {
  (void)in_sizes; (void)n_in; (void)out_size; (void)ws_size;
  const int* s = (const int*)d_in[0];
  const float* x = (const float*)d_in[1];
  const int* src = (const int*)d_in[2];
  const int* dst = (const int*)d_in[3];
  const float* target = (const float*)d_in[4];
  const float* emb = (const float*)d_in[5];
  auto P = [&](int i) { return (const float*)d_in[i]; };

  // ---- workspace layout (~197 MB) ----
  float* base = (float*)d_ws;
  size_t off = 0;
  auto alloc = [&](size_t n) { float* p = base + off; off += n; return p; };
  float* w = alloc((size_t)NE * 64);
  float* rlen = alloc(NE);
  float* Y1 = alloc((size_t)NE * 3);
  float* Y2 = alloc((size_t)NE * 5);
  float* msg = alloc((size_t)NE * MSGW);
  float* logits = alloc((size_t)NE * 4);
  float* nodeMax = alloc((size_t)NN * 4);
  float* den = alloc((size_t)NN * 4);
  float* qnode = alloc((size_t)NN * 32);
  float* xbuf = alloc((size_t)NN * 3);
  float* hA0 = alloc((size_t)NN * H0S);
  float* hA1 = alloc((size_t)NN * H1S);
  float* hA2 = alloc((size_t)NN * H2S);
  float* hB0 = alloc((size_t)NN * H0S);
  float* hB1 = alloc((size_t)NN * H1S);
  float* hB2 = alloc((size_t)NN * H2S);
  float* lossAcc = alloc(4);
  // f16 fragment areas, 32B aligned
  auto allocH = [&](size_t nh) {
    off = (off + 7) & ~(size_t)7;
    _Float16* p = (_Float16*)(base + off);
    off += (nh + 1) / 2;
    return p;
  };
  // layer index: 0 = attn layer0, 1 = attn layer1, 2 = final gconv
  int Rt[3] = {7, 9, 13};       // rout n-tiles
  int SCt[3] = {4, 4, 5};       // sc n-tiles
  int kch[3] = {2, 1, 1};       // sc k-chunks (fin0/32)
  _Float16 *pk_rw1[3], *pk_rw2[3], *pk_sc[3];
  for (int i = 0; i < 3; ++i) {
    pk_rw1[i] = allocH(6 * 512);
    pk_rw2[i] = allocH((size_t)Rt[i] * 512);
    pk_sc[i]  = allocH((size_t)SCt[i] * kch[i] * 512);
  }

  auto FILL = [&](float* p, int n, float v) {
    fill_kernel<<<(n + 255) / 256, 256, 0, stream>>>(p, n, v);
  };
  auto swp = [](float*& a, float*& b) { float* t = a; a = b; b = t; };
  auto launch_pack = [&](PackCfg pc) {
    int total = pc.ntiles * pc.kchunks * 32;
    pack_kernel<<<(total + 63) / 64, 64, 0, stream>>>(pc);
  };

  // ---- pre-pack weight fragments (once per call, reused by 6 edge passes) ----
  {
    const float* rw1s[3] = {P(P_L0_rw1), P(P_L1_rw1), P(P_F_rw1)};
    const float* rw2s[3] = {P(P_L0_rw2), P(P_L1_rw2), P(P_F_rw2)};
    const float* W00s[3] = {P(P_L0_W00), P(P_L1_W00), P(P_F_W00)};
    const float* W01s[3] = {P(P_L0_W01), P(P_L1_W01), P(P_F_W01)};
    const float* W02s[3] = {P(P_L0_W02), P(P_L1_W02), nullptr};
    int Rs[3] = {100, 136, 196};
    int fin0s[3] = {64, 32, 32};
    int mo0s[3] = {32, 32, 64}, mo1s[3] = {16, 16, 2}, mo2s[3] = {4, 4, 0};
    for (int i = 0; i < 3; ++i) {
      PackCfg a{};
      a.mode = 0; a.B = rw1s[i]; a.ld = 32; a.kmax = 65; a.nmax = 32;
      a.ntiles = 2; a.kchunks = 3; a.out = pk_rw1[i];
      launch_pack(a);
      PackCfg b{};
      b.mode = 0; b.B = rw2s[i]; b.ld = Rs[i]; b.kmax = 32; b.nmax = Rs[i];
      b.ntiles = Rt[i]; b.kchunks = 1; b.out = pk_rw2[i];
      launch_pack(b);
      PackCfg cc{};
      cc.mode = 1; cc.W00 = W00s[i]; cc.W01 = W01s[i]; cc.W02 = W02s[i];
      cc.fin0 = fin0s[i]; cc.mo0 = mo0s[i]; cc.mo1 = mo1s[i]; cc.mo2 = mo2s[i];
      cc.ntiles = SCt[i]; cc.kchunks = kch[i]; cc.out = pk_sc[i];
      launch_pack(cc);
    }
  }

  FILL(lossAcc, 4, 0.f);
  init_kernel<<<NN / 256, 256, 0, stream>>>(s, x, emb, hA0, hA1, hA2, xbuf);
  w_kernel<<<(NE * 64) / 256, 256, 0, stream>>>(src, dst, hA0, w);

  float *h0c = hA0, *h1c = hA1, *h2c = hA2;
  float *h0n = hB0, *h1n = hB1, *h2n = hB2;

  struct LP {
    const float *rb1, *rb2, *W10, *W11, *W20, *W22, *Wq, *S0, *S1, *S2;
    const float *a0, *b0, *a1, *b1, *a2, *b2;
    int fin0, C1, C2, R, SC;
    int off00, off10, off20, off01, off11, off02, off22;
  };
  LP L0{};
  L0.rb1 = P(P_L0_rb1); L0.rb2 = P(P_L0_rb2);
  L0.W10 = P(P_L0_W10); L0.W11 = P(P_L0_W11); L0.W20 = nullptr; L0.W22 = nullptr;
  L0.Wq = P(P_L0_Wq); L0.S0 = P(P_L0_S0); L0.S1 = P(P_L0_S1); L0.S2 = nullptr;
  L0.a0 = P(P_L0_a0); L0.b0 = P(P_L0_b0); L0.a1 = P(P_L0_a1); L0.b1 = P(P_L0_b1);
  L0.a2 = P(P_L0_a2); L0.b2 = P(P_L0_b2);
  L0.fin0 = 64; L0.C1 = 2; L0.C2 = 0; L0.R = 100; L0.SC = 52;
  L0.off00 = 0; L0.off10 = 32; L0.off20 = -1; L0.off01 = 64; L0.off11 = 80;
  L0.off02 = 96; L0.off22 = -1;

  LP L1{};
  L1.rb1 = P(P_L1_rb1); L1.rb2 = P(P_L1_rb2);
  L1.W10 = P(P_L1_W10); L1.W11 = P(P_L1_W11); L1.W20 = P(P_L1_W20); L1.W22 = P(P_L1_W22);
  L1.Wq = P(P_L1_Wq); L1.S0 = P(P_L1_S0); L1.S1 = P(P_L1_S1); L1.S2 = P(P_L1_S2);
  L1.a0 = P(P_L1_a0); L1.b0 = P(P_L1_b0); L1.a1 = P(P_L1_a1); L1.b1 = P(P_L1_b1);
  L1.a2 = P(P_L1_a2); L1.b2 = P(P_L1_b2);
  L1.fin0 = 32; L1.C1 = 16; L1.C2 = 4; L1.R = 136; L1.SC = 52;
  L1.off00 = 0; L1.off10 = 32; L1.off20 = 64; L1.off01 = 96; L1.off11 = 112;
  L1.off02 = 128; L1.off22 = 132;

  LP* LL[2] = {&L0, &L1};

  for (int iter = 0; iter < 2; ++iter) {
    geom_kernel<<<NE / 256, 256, 0, stream>>>(src, dst, xbuf, rlen, Y1, Y2);
    for (int l = 0; l < 2; ++l) {
      LP& lp = *LL[l];
      q_kernel<<<(NN * 32) / 256, 256, 0, stream>>>(h0c, lp.Wq, lp.fin0, qnode);
      FILL(nodeMax, NN * 4, -1e30f);
      FILL(den, NN * 4, 0.f);
      FILL(h0n, NN * H0S, 0.f);
      FILL(h1n, NN * H1S, 0.f);
      FILL(h2n, NN * H2S, 0.f);

      EdgeCfg ec{};
      ec.src = src; ec.dst = dst; ec.w = w; ec.rlen = rlen; ec.Y1 = Y1; ec.Y2 = Y2;
      ec.h0 = h0c; ec.h1 = h1c; ec.h2 = h2c; ec.qnode = qnode;
      ec.rb1 = lp.rb1; ec.rb2 = lp.rb2;
      ec.rw1p = pk_rw1[l]; ec.rw2p = pk_rw2[l]; ec.scp = pk_sc[l];
      ec.W10 = lp.W10; ec.W11 = lp.W11; ec.W20 = lp.W20; ec.W22 = lp.W22;
      ec.msg = msg; ec.logits = logits; ec.nodeMax = nodeMax;
      ec.agg0 = nullptr; ec.agg1 = nullptr;
      ec.fin0 = lp.fin0; ec.C1 = lp.C1; ec.C2 = lp.C2;
      ec.mo0 = 32; ec.mo1 = 16; ec.mo2 = 4;
      ec.R = lp.R; ec.SC = lp.SC;
      ec.off00 = lp.off00; ec.off10 = lp.off10; ec.off20 = lp.off20;
      ec.off01 = lp.off01; ec.off11 = lp.off11; ec.off02 = lp.off02;
      ec.off22 = lp.off22;
      ec.attn = 1;
      edge_kernel<<<NE / 32, 64, 0, stream>>>(ec);

      den_kernel<<<(NE * 4) / 256, 256, 0, stream>>>(dst, logits, nodeMax, den);
      attn_scatter_kernel<<<(NE * 4) / 256, 256, 0, stream>>>(
          dst, logits, nodeMax, den, msg, h0n, h1n, h2n);

      NodeCfg nc{};
      nc.agg0 = h0n; nc.agg1 = h1n; nc.agg2 = h2n;
      nc.h0 = h0c; nc.h1 = h1c; nc.h2 = h2c;
      nc.S0 = lp.S0; nc.S1 = lp.S1; nc.S2 = lp.S2;
      nc.a0 = lp.a0; nc.b0 = lp.b0; nc.a1 = lp.a1; nc.b1 = lp.b1;
      nc.a2 = lp.a2; nc.b2 = lp.b2;
      nc.o0 = h0n; nc.o1 = h1n; nc.o2 = h2n;
      nc.fin0 = lp.fin0; nc.C1 = lp.C1; nc.C2 = lp.C2;
      node_attn_kernel<<<NN / 256, 256, 0, stream>>>(nc);

      swp(h0c, h0n); swp(h1c, h1n); swp(h2c, h2n);
    }

    // final gconv: direct atomic scatter (no per-edge message buffer)
    FILL(h0n, NN * H0S, 0.f);
    FILL(h1n, NN * H1S, 0.f);
    EdgeCfg ef{};
    ef.src = src; ef.dst = dst; ef.w = w; ef.rlen = rlen; ef.Y1 = Y1; ef.Y2 = Y2;
    ef.h0 = h0c; ef.h1 = h1c; ef.h2 = h2c; ef.qnode = nullptr;
    ef.rb1 = P(P_F_rb1); ef.rb2 = P(P_F_rb2);
    ef.rw1p = pk_rw1[2]; ef.rw2p = pk_rw2[2]; ef.scp = pk_sc[2];
    ef.W10 = P(P_F_W10); ef.W11 = P(P_F_W11); ef.W20 = P(P_F_W20); ef.W22 = nullptr;
    ef.msg = nullptr; ef.logits = nullptr; ef.nodeMax = nullptr;
    ef.agg0 = h0n; ef.agg1 = h1n;
    ef.fin0 = 32; ef.C1 = 16; ef.C2 = 4;
    ef.mo0 = 64; ef.mo1 = 2; ef.mo2 = 0;
    ef.R = 196; ef.SC = 66;
    ef.off00 = 0; ef.off10 = 64; ef.off20 = 128; ef.off01 = 192; ef.off11 = 194;
    ef.off02 = -1; ef.off22 = -1;
    ef.attn = 0;
    edge_kernel<<<NE / 32, 64, 0, stream>>>(ef);

    node_final_kernel<<<NN / 256, 256, 0, stream>>>(
        h0n, h1n, h0c, h1c, P(P_F_Ws0), P(P_F_Ws1), h0n, h1n, xbuf);
    swp(h0c, h0n); swp(h1c, h1n); swp(h2c, h2n);
  }

  out_kernel<<<NN / 256, 256, 0, stream>>>(h1c, target, (float*)d_out, lossAcc);
  loss_kernel<<<1, 4, 0, stream>>>(lossAcc, (float*)d_out);
}
